// AnomalyScores_71150428226180
// MI455X (gfx1250) — matrix operand order repaired and checked by reading, not compiled, not run on hardware
//
#include <hip/hip_runtime.h>
#include <hip/hip_bf16.h>
#include <stdint.h>

// ---------------------------------------------------------------------------
// PatchCore anomaly score, CDNA5 (gfx1250) version.
// argmin_n ||a_m - b_n||  ==  argmin_n (||b_n||^2 - 2 a_m.b_n) computed as a
// split-bf16 WMMA GEMM fused with the row min/argmin. B (coreset) tiles are
// staged into LDS with double-buffered global_load_async_to_lds_b128 and
// shared by two M-tile waves, halving L2 traffic vs direct VGPR loads.
// A tiles are explicitly preloaded into registers (192 VGPRs / wave).
// ---------------------------------------------------------------------------

typedef __attribute__((ext_vector_type(16))) __bf16 v16bf;
typedef __attribute__((ext_vector_type(8)))  float  v8f;

#define DIMK   384
#define KSTEPS (DIMK / 32)   // 12
#define MROWS  6272          // B*P = 8*784
#define NCORE  16384
#define BATCH  8
#define PPATCH 784
#define KNN    9

#define NSTRIPS         4                       // N strips per block
#define STRIP_COLS      (NCORE / NSTRIPS)       // 4096
#define TILES_PER_STRIP (STRIP_COLS / 16)       // 256
#define TILE_ELEMS      (16 * DIMK)             // ushorts per 16-row tile (12KB)
#define BUF_STRIDE      (NSTRIPS * 2 * TILE_ELEMS)  // elements per LDS buffer

union V16U {
    v16bf v;
    uint4 q[2];
};

__device__ __forceinline__ unsigned short f32_to_bf16_rne(float x) {
    unsigned int u = __float_as_uint(x);
    unsigned int r = u + 0x7FFFu + ((u >> 16) & 1u);
    return (unsigned short)(r >> 16);
}
__device__ __forceinline__ float bf16_bits_to_f32(unsigned short h) {
    return __uint_as_float(((unsigned int)h) << 16);
}

__device__ __forceinline__ void wait_asynccnt0() {
#if __has_builtin(__builtin_amdgcn_s_wait_asynccnt)
    __builtin_amdgcn_s_wait_asynccnt(0);
#else
    asm volatile("s_wait_asynccnt 0x0" ::: "memory");
#endif
}

// Async-copy one contiguous 12KB B tile (16 rows x 384 bf16) global -> LDS.
// 32 lanes x b128 = 512B per instruction, 24 instructions.
__device__ __forceinline__ void async_copy_tile(const unsigned short* __restrict__ g,
                                                const unsigned short* lds_dst, int lane) {
    const char* gp   = (const char*)g + lane * 16;
    unsigned int lof = (unsigned int)(uintptr_t)lds_dst + (unsigned int)(lane * 16);
    #pragma unroll
    for (int i = 0; i < 24; ++i) {
        asm volatile("global_load_async_to_lds_b128 %0, %1, off"
                     :: "v"(lof), "v"(gp)
                     : "memory");
        gp  += 512;
        lof += 512;
    }
}

// ---------------------------------------------------------------------------
// Kernel 1: f32 -> (bf16 hi, bf16 lo) split + per-row squared norms.
// One wave per row of 384 elements.
// ---------------------------------------------------------------------------
__global__ __launch_bounds__(256) void k_split_bf16(
    const float* __restrict__ src,
    unsigned short* __restrict__ hi,
    unsigned short* __restrict__ lo,
    float* __restrict__ norms,
    int rows)
{
    const int wave = threadIdx.x >> 5;
    const int lane = threadIdx.x & 31;
    const int row  = blockIdx.x * 8 + wave;
    if (row >= rows) return;

    const float* s = src + (size_t)row * DIMK;
    unsigned short* h = hi + (size_t)row * DIMK;
    unsigned short* l = lo + (size_t)row * DIMK;

    float sq = 0.0f;
    #pragma unroll
    for (int d = lane; d < DIMK; d += 32) {
        float x = s[d];
        unsigned short hb = f32_to_bf16_rne(x);
        float hf = bf16_bits_to_f32(hb);
        h[d] = hb;
        l[d] = f32_to_bf16_rne(x - hf);
        sq = fmaf(x, x, sq);
    }
    #pragma unroll
    for (int m = 1; m < 32; m <<= 1) sq += __shfl_xor(sq, m, 32);
    if (lane == 0) norms[row] = sq;
}

// ---------------------------------------------------------------------------
// Kernel 2: fused GEMM (split-bf16, 3 WMMAs per K-step) + row min/argmin.
// grid.x = MROWS/32; block = 8 waves = (2 M-tiles) x (4 N-strips).
// Wave w: strip s = w&3, M-tile sub = w>>2. The two waves of a strip share
// the B tile staged in LDS by the sub==0 wave (double buffered, ASYNCcnt).
// ---------------------------------------------------------------------------
__global__ __launch_bounds__(256) void k_nn_min(
    const unsigned short* __restrict__ Ah, const unsigned short* __restrict__ Al,
    const unsigned short* __restrict__ Bh, const unsigned short* __restrict__ Bl,
    const float* __restrict__ a2, const float* __restrict__ b2,
    float* __restrict__ patch_scores, int* __restrict__ locations)
{
    extern __shared__ __align__(16) unsigned short ldsB[]; // [2][NSTRIPS][2][TILE_ELEMS] = 192KB

    const int tid  = threadIdx.x;
    const int wave = tid >> 5;
    const int lane = tid & 31;
    const int r    = lane & 15;   // row-within-tile for A, col-within-tile for B/C
    const int h    = lane >> 4;   // half-wave
    const int s    = wave & 3;    // N strip
    const int sub  = wave >> 2;   // M tile within block
    const int m0   = (blockIdx.x * 2 + sub) * 16;
    const int stripBase = s * STRIP_COLS;

    // LDS element offsets of this strip's hi/lo tiles within one buffer
    const unsigned tH = (unsigned)((s * 2 + 0) * TILE_ELEMS);
    const unsigned tL = (unsigned)((s * 2 + 1) * TILE_ELEMS);

    // ---- explicit A-tile preload: 12 K-steps x (hi,lo) kept in registers ----
    // A 16x32 bf16 layout: half-wave h holds K octets [8h..8h+7] (v0-3) and
    // [16+8h..16+8h+7] (v4-7).
    const unsigned short* aHrow = Ah + (size_t)(m0 + r) * DIMK;
    const unsigned short* aLrow = Al + (size_t)(m0 + r) * DIMK;
    V16U aTH[KSTEPS], aTL[KSTEPS];
    #pragma unroll
    for (int k = 0; k < KSTEPS; ++k) {
        const int kb = k * 32;
        aTH[k].q[0] = *(const uint4*)(aHrow + kb + h * 8);
        aTH[k].q[1] = *(const uint4*)(aHrow + kb + 16 + h * 8);
        aTL[k].q[0] = *(const uint4*)(aLrow + kb + h * 8);
        aTL[k].q[1] = *(const uint4*)(aLrow + kb + 16 + h * 8);
    }

    float runMin[8];
    int   runIdx[8];
    #pragma unroll
    for (int j = 0; j < 8; ++j) { runMin[j] = 3.0e38f; runIdx[j] = 0; }

    // prologue: stage tile 0 into buffer 0
    if (sub == 0) {
        async_copy_tile(Bh + (size_t)stripBase * DIMK, ldsB + tH, lane);
        async_copy_tile(Bl + (size_t)stripBase * DIMK, ldsB + tL, lane);
        wait_asynccnt0();
    }
    __syncthreads();

    int buf = 0;
    for (int it = 0; it < TILES_PER_STRIP; ++it) {
        // stage next tile into the other buffer while computing on this one
        if (sub == 0 && (it + 1) < TILES_PER_STRIP) {
            const size_t nn0 = (size_t)(stripBase + (it + 1) * 16) * DIMK;
            const unsigned obo = buf ? 0u : (unsigned)BUF_STRIDE;
            async_copy_tile(Bh + nn0, ldsB + obo + tH, lane);
            async_copy_tile(Bl + nn0, ldsB + obo + tL, lane);
        }

        const int      n0 = stripBase + it * 16;
        const unsigned bo = buf ? (unsigned)BUF_STRIDE : 0u;   // integer select,
                                                               // keeps AS3 provenance
        v8f c = {0.f, 0.f, 0.f, 0.f, 0.f, 0.f, 0.f, 0.f};

        #pragma unroll
        for (int k = 0; k < KSTEPS; ++k) {
            const int kb = k * 32;
            // B 32x16 bf16 layout: half-wave h holds K octets [8h..8h+7] and
            // [16+8h..16+8h+7]; fragments read from the LDS-staged tile (ds_load_b128).
            V16U bh, bl;
            bh.q[0] = *(const uint4*)(ldsB + bo + tH + r * DIMK + kb + h * 8);
            bh.q[1] = *(const uint4*)(ldsB + bo + tH + r * DIMK + kb + 16 + h * 8);
            bl.q[0] = *(const uint4*)(ldsB + bo + tL + r * DIMK + kb + h * 8);
            bl.q[1] = *(const uint4*)(ldsB + bo + tL + r * DIMK + kb + 16 + h * 8);

            // split-bf16: hi*hi + hi*lo + lo*hi (drop lo*lo) ~ fp32 accuracy
            c = __builtin_amdgcn_wmma_f32_16x16x32_bf16(false, aTH[k].v, false, bh.v,
                                                        (short)0, c, false, false);
            c = __builtin_amdgcn_wmma_f32_16x16x32_bf16(false, aTH[k].v, false, bl.v,
                                                        (short)0, c, false, false);
            c = __builtin_amdgcn_wmma_f32_16x16x32_bf16(false, aTL[k].v, false, bh.v,
                                                        (short)0, c, false, false);
        }

        const int   n  = n0 + r;
        const float bn = b2[n];
        #pragma unroll
        for (int j = 0; j < 8; ++j) {
            float d = bn - 2.0f * c[j];     // ||b||^2 - 2 a.b (row m0+j+8h, col n)
            if (d < runMin[j]) { runMin[j] = d; runIdx[j] = n; }
        }

        if (sub == 0) wait_asynccnt0();     // next-buffer copies complete
        __syncthreads();                    // publish LDS / protect old buffer
        buf ^= 1;
    }

    // Cross-lane argmin over the 16 columns held by each half-wave.
    #pragma unroll
    for (int j = 0; j < 8; ++j) {
        float v = runMin[j];
        int   i = runIdx[j];
        #pragma unroll
        for (int m = 1; m < 16; m <<= 1) {
            float ov = __shfl_xor(v, m, 32);
            int   oi = __shfl_xor(i, m, 32);
            if (ov < v || (ov == v && oi < i)) { v = ov; i = oi; }
        }
        runMin[j] = v; runIdx[j] = i;
    }

    __shared__ float sMin[8][16];
    __shared__ int   sIdx[8][16];
    if (r == 0) {
        #pragma unroll
        for (int j = 0; j < 8; ++j) {
            sMin[wave][8 * h + j] = runMin[j];
            sIdx[wave][8 * h + j] = runIdx[j];
        }
    }
    __syncthreads();

    // combine the 4 strips of each M-tile
    if (tid < 32) {
        const int sb  = tid >> 4;       // which M tile
        const int row = tid & 15;
        float v = sMin[sb * 4 + 0][row];
        int   i = sIdx[sb * 4 + 0][row];
        #pragma unroll
        for (int w = 1; w < 4; ++w) {
            float ov = sMin[sb * 4 + w][row];
            int   oi = sIdx[sb * 4 + w][row];
            if (ov < v || (ov == v && oi < i)) { v = ov; i = oi; }
        }
        const int grow = (blockIdx.x * 2 + sb) * 16 + row;
        patch_scores[grow] = sqrtf(fmaxf(a2[grow] + v, 0.0f));
        locations[grow]    = i;
    }
}

// ---------------------------------------------------------------------------
// Kernel 3: one block per batch image. argmax patch -> kNN support set ->
// softmax reweighting -> final score. All tiny relative to kernel 2.
// ---------------------------------------------------------------------------
__global__ __launch_bounds__(256) void k_final(
    const float* __restrict__ emb,        // original f32 [B*P, D]
    const float* __restrict__ core,       // original f32 [N, D]
    const float* __restrict__ a2, const float* __restrict__ b2,
    const float* __restrict__ patch_scores, const int* __restrict__ locations,
    float* __restrict__ out)
{
    const int b   = blockIdx.x;
    const int tid = threadIdx.x;

    __shared__ float rv[256];
    __shared__ int   ri[256];
    __shared__ int   rs[256];

    // -------- phase 1: argmax over the 784 patch scores --------
    float best = -3.0e38f; int bi = 0x7FFFFFFF;
    for (int p = tid; p < PPATCH; p += 256) {
        float s = patch_scores[b * PPATCH + p];
        if (s > best || (s == best && p < bi)) { best = s; bi = p; }
    }
    rv[tid] = best; ri[tid] = bi;
    __syncthreads();
    for (int s = 128; s > 0; s >>= 1) {
        if (tid < s) {
            if (rv[tid + s] > rv[tid] || (rv[tid + s] == rv[tid] && ri[tid + s] < ri[tid])) {
                rv[tid] = rv[tid + s]; ri[tid] = ri[tid + s];
            }
        }
        __syncthreads();
    }
    __shared__ int   s_maxp, s_nn;
    __shared__ float s_score;
    if (tid == 0) {
        s_maxp  = ri[0];
        s_score = rv[0];
        s_nn    = locations[b * PPATCH + ri[0]];
    }
    __syncthreads();

    __shared__ float s_nnv[DIMK];
    __shared__ float s_maxf[DIMK];
    {
        const float* maxf = emb  + (size_t)(b * PPATCH + s_maxp) * DIMK;
        const float* nnv  = core + (size_t)s_nn * DIMK;
        for (int i = tid; i < DIMK; i += 256) { s_nnv[i] = nnv[i]; s_maxf[i] = maxf[i]; }
    }
    __syncthreads();

    // -------- phase 2: per-thread top-9 of dist(nn_sample, coreset) --------
    float tv[KNN]; int ti[KNN];
    #pragma unroll
    for (int k = 0; k < KNN; ++k) { tv[k] = 3.0e38f; ti[k] = 0x7FFFFFFF; }
    const float nn2 = b2[s_nn];
    for (int n = tid; n < NCORE; n += 256) {
        const float* c = core + (size_t)n * DIMK;
        float dot = 0.0f;
        for (int d = 0; d < DIMK; ++d) dot = fmaf(s_nnv[d], c[d], dot);
        float dist = sqrtf(fmaxf(nn2 + b2[n] - 2.0f * dot, 0.0f));
        if (dist < tv[KNN - 1] || (dist == tv[KNN - 1] && n < ti[KNN - 1])) {
            tv[KNN - 1] = dist; ti[KNN - 1] = n;
            #pragma unroll
            for (int k = KNN - 1; k > 0; --k) {
                bool sw = (tv[k] < tv[k - 1]) || (tv[k] == tv[k - 1] && ti[k] < ti[k - 1]);
                if (sw) {
                    float fv = tv[k]; tv[k] = tv[k - 1]; tv[k - 1] = fv;
                    int   iv = ti[k]; ti[k] = ti[k - 1]; ti[k - 1] = iv;
                }
            }
        }
    }

    // -------- phase 3: merge 256x9 candidates -> global top-9 --------
    __shared__ float candV[256 * KNN];
    __shared__ int   candI[256 * KNN];
    #pragma unroll
    for (int k = 0; k < KNN; ++k) { candV[tid * KNN + k] = tv[k]; candI[tid * KNN + k] = ti[k]; }
    __syncthreads();

    __shared__ int s_sup[KNN];
    for (int round = 0; round < KNN; ++round) {
        float lv = 3.0e38f; int li = 0x7FFFFFFF; int ls = 0;
        #pragma unroll
        for (int k = 0; k < KNN; ++k) {
            float v = candV[tid * KNN + k]; int ix = candI[tid * KNN + k];
            if (v < lv || (v == lv && ix < li)) { lv = v; li = ix; ls = tid * KNN + k; }
        }
        rv[tid] = lv; ri[tid] = li; rs[tid] = ls;
        __syncthreads();
        for (int s = 128; s > 0; s >>= 1) {
            if (tid < s) {
                if (rv[tid + s] < rv[tid] || (rv[tid + s] == rv[tid] && ri[tid + s] < ri[tid])) {
                    rv[tid] = rv[tid + s]; ri[tid] = ri[tid + s]; rs[tid] = rs[tid + s];
                }
            }
            __syncthreads();
        }
        if (tid == 0) {
            s_sup[round] = ri[0];
            candV[rs[0]] = 3.0e38f;
            candI[rs[0]] = 0x7FFFFFFF;
        }
        __syncthreads();
    }

    // -------- phase 4: distances max_patch -> support, softmax weight --------
    __shared__ float s_dsup[KNN];
    if (tid < KNN) {
        const int idx = s_sup[tid];
        const float* s = core + (size_t)idx * DIMK;
        float dot = 0.0f;
        for (int d = 0; d < DIMK; ++d) dot = fmaf(s_maxf[d], s[d], dot);
        const float mf2 = a2[b * PPATCH + s_maxp];
        s_dsup[tid] = sqrtf(fmaxf(mf2 + b2[idx] - 2.0f * dot, 0.0f));
    }
    __syncthreads();
    if (tid == 0) {
        float mx = -3.0e38f;
        for (int k = 0; k < KNN; ++k) mx = fmaxf(mx, s_dsup[k]);
        float sum = 0.0f;
        for (int k = 0; k < KNN; ++k) sum += expf(s_dsup[k] - mx);
        float w = 1.0f - expf(s_dsup[0] - mx) / sum;
        out[b] = w * s_score;
    }
}

// ---------------------------------------------------------------------------
extern "C" void kernel_launch(void* const* d_in, const int* in_sizes, int n_in,
                              void* d_out, int out_size, void* d_ws, size_t ws_size,
                              hipStream_t stream)
{
    (void)in_sizes; (void)n_in; (void)out_size; (void)ws_size;

    const float* emb  = (const float*)d_in[0];   // [6272, 384] f32
    const float* core = (const float*)d_in[1];   // [16384, 384] f32
    float* out = (float*)d_out;                  // [8] f32

    // workspace layout (256-byte aligned slices)
    char* ws = (char*)d_ws;
    size_t off = 0;
    auto take = [&](size_t bytes) -> char* {
        char* p = ws + off;
        off = (off + bytes + 255) & ~(size_t)255;
        return p;
    };
    unsigned short* Ah = (unsigned short*)take((size_t)MROWS * DIMK * 2);
    unsigned short* Al = (unsigned short*)take((size_t)MROWS * DIMK * 2);
    unsigned short* Bh = (unsigned short*)take((size_t)NCORE * DIMK * 2);
    unsigned short* Bl = (unsigned short*)take((size_t)NCORE * DIMK * 2);
    float* a2  = (float*)take((size_t)MROWS * 4);
    float* b2  = (float*)take((size_t)NCORE * 4);
    float* ps  = (float*)take((size_t)MROWS * 4);
    int*   loc = (int*)  take((size_t)MROWS * 4);

    // 1) split-bf16 conversion + norms
    hipLaunchKernelGGL(k_split_bf16, dim3(MROWS / 8), dim3(256), 0, stream,
                       emb, Ah, Al, a2, MROWS);
    hipLaunchKernelGGL(k_split_bf16, dim3(NCORE / 8), dim3(256), 0, stream,
                       core, Bh, Bl, b2, NCORE);

    // 2) fused WMMA GEMM + min/argmin; 196 blocks x (2 M-tiles, 4 N-strips),
    //    B tiles double-buffered in LDS via async copies (192KB dynamic LDS).
    const size_t ldsBytes = (size_t)2 * BUF_STRIDE * sizeof(unsigned short);
    hipLaunchKernelGGL(k_nn_min, dim3(MROWS / 32), dim3(256), ldsBytes, stream,
                       Ah, Al, Bh, Bl, a2, b2, ps, loc);

    // 3) per-batch argmax + kNN support + softmax reweight
    hipLaunchKernelGGL(k_final, dim3(BATCH), dim3(256), 0, stream,
                       emb, core, a2, b2, ps, loc, out);
}
